// MPNN_61830349193837
// MI455X (gfx1250) — matrix operand order repaired and checked by reading, not compiled, not run on hardware
//
#include <hip/hip_runtime.h>
#include <hip/hip_bf16.h>
#include <stdint.h>

// ---- problem constants (fixed by reference) ----
#define ATOM_FDIM 133
#define BOND_FDIM 147
#define HID       300
#define NB_ATOMS  200000
#define NB_BONDS  400000
#define MAX_NB    6
#define NB_MOLS   10000
#define NPAD      384      // 24 N-tiles of 16 (zero padded weights -> unclamped B loads)
// K paddings (multiples of 64)
#define KP_I      192
#define KP_H      320
#define KP_O      448

typedef __attribute__((ext_vector_type(16))) __bf16 v16bf;
typedef __attribute__((ext_vector_type(8)))  float  v8f;

union BFrag { v16bf v; uint4 q[2]; unsigned short s[16]; };
union FAcc  { v8f   v; float f[8]; };

__device__ __forceinline__ unsigned short f2bf(float f) {
  unsigned int u = __float_as_uint(f);
  u += 0x7FFFu + ((u >> 16) & 1u);          // round-to-nearest-even
  return (unsigned short)(u >> 16);
}
__device__ __forceinline__ float bf2f(unsigned short s) {
  return __uint_as_float(((unsigned int)s) << 16);
}
__device__ __forceinline__ unsigned int pk2bf(float a, float b) {
  return (unsigned int)f2bf(a) | ((unsigned int)f2bf(b) << 16);
}

// ---- utility kernels ----
__global__ void fill_zero_k(float* p, int n) {
  int i = blockIdx.x * 256 + threadIdx.x;
  if (i < n) p[i] = 0.0f;
}

// pack W[K][N] fp32 -> Wt[NPAD][Kpad] bf16 (N-major, zero padded both dims)
__global__ void pack_w_k(const float* __restrict__ W, int K, int N,
                         unsigned short* __restrict__ Wt, int Kpad) {
  int i = blockIdx.x * 256 + threadIdx.x;
  if (i >= Kpad * NPAD) return;
  int n = i / Kpad, k = i % Kpad;
  float v = W[(size_t)min(k, K - 1) * N + min(n, N - 1)];
  v = (k < K && n < N) ? v : 0.0f;
  Wt[(size_t)n * Kpad + k] = f2bf(v);
}

// f_bonds fp32 [M x 147] -> bf16 [M x 192] row-padded   (96 thr: 1 pair each)
__global__ void prep_bonds_k(const float* __restrict__ fb,
                             unsigned short* __restrict__ A) {
  int r = blockIdx.x;
  int t = threadIdx.x;
  const float* src = fb + (size_t)r * BOND_FDIM;
  unsigned int* dst = (unsigned int*)(A + (size_t)r * KP_I);
  int k0 = 2 * t, k1 = 2 * t + 1;
  float v0 = src[min(k0, BOND_FDIM - 1)];
  float v1 = src[min(k1, BOND_FDIM - 1)];
  v0 = (k0 < BOND_FDIM) ? v0 : 0.0f;
  v1 = (k1 < BOND_FDIM) ? v1 : 0.0f;
  dst[t] = pk2bf(v0, v1);
}

// a_message[a][h] = sum_j bf2f(msg[a2b[a][j]][h])   (160 thr: 1 pair each)
__global__ void amsg_k(const unsigned short* __restrict__ msg,
                       const int* __restrict__ a2b,
                       float* __restrict__ amsg) {
  int a = blockIdx.x;
  int p = threadIdx.x;
  if (p >= HID / 2) return;
  int nb[MAX_NB];
#pragma unroll
  for (int j = 0; j < MAX_NB; ++j) nb[j] = a2b[(size_t)a * MAX_NB + j];
  float s0 = 0.0f, s1 = 0.0f;
#pragma unroll
  for (int j = 0; j < MAX_NB; ++j) {
    unsigned int m2 = ((const unsigned int*)(msg + (size_t)nb[j] * HID))[p];
    s0 += bf2f((unsigned short)(m2 & 0xFFFFu));
    s1 += bf2f((unsigned short)(m2 >> 16));
  }
  ((float2*)(amsg + (size_t)a * HID))[p] = make_float2(s0, s1);
}

// new_msg[b][k] = amsg[b2a[b]][k] - msg[b2revb[b]][k]  -> bf16 [M x 320]  (160 thr)
__global__ void prep_newmsg_k(const float* __restrict__ amsg,
                              const unsigned short* __restrict__ msg,
                              const int* __restrict__ b2a,
                              const int* __restrict__ b2revb,
                              unsigned short* __restrict__ A) {
  int r = blockIdx.x;
  int p = threadIdx.x;                      // 0..159 pairs
  int ba = b2a[r], br = b2revb[r];
  unsigned int o = 0u;
  if (p < HID / 2) {
    float2 a2 = ((const float2*)(amsg + (size_t)ba * HID))[p];
    unsigned int m2 = ((const unsigned int*)(msg + (size_t)br * HID))[p];
    o = pk2bf(a2.x - bf2f((unsigned short)(m2 & 0xFFFFu)),
              a2.y - bf2f((unsigned short)(m2 >> 16)));
  }
  ((unsigned int*)(A + (size_t)r * KP_H))[p] = o;
}

// [f_atoms | a_message] -> bf16 [M x 448]   (224 thr: 1 pair each)
__global__ void prep_atoms_k(const float* __restrict__ fa,
                             const float* __restrict__ amsg,
                             unsigned short* __restrict__ A) {
  int r = blockIdx.x;
  int p = threadIdx.x;                      // 0..223 pairs
  const float* pf = fa + (size_t)r * ATOM_FDIM;
  const float* pm = amsg + (size_t)r * HID;
  int e0 = 2 * p, e1 = 2 * p + 1;
  float f0 = pf[min(e0, ATOM_FDIM - 1)];
  float f1 = pf[min(e1, ATOM_FDIM - 1)];
  float m0 = pm[min(max(e0 - ATOM_FDIM, 0), HID - 1)];
  float m1 = pm[min(max(e1 - ATOM_FDIM, 0), HID - 1)];
  float v0 = (e0 < ATOM_FDIM) ? f0 : ((e0 < ATOM_FDIM + HID) ? m0 : 0.0f);
  float v1 = (e1 < ATOM_FDIM) ? f1 : ((e1 < ATOM_FDIM + HID) ? m1 : 0.0f);
  ((unsigned int*)(A + (size_t)r * KP_O))[p] = pk2bf(v0, v1);
}

// ---- WMMA GEMM core: O[M x 300] = act(Abf[M x KPAD] @ Wt^T + C) ----
// CMODE 0: C=0   1: C=inp tile   2: C=bias row
// OMODE 0: store inp fp32 + relu->msg bf16   1: relu->msg bf16   2: relu->hid fp32
// 128 thr = 4 waves; wave owns 2 N-tiles; K fully unrolled (KPAD compile-time):
// 4 WMMA per barrier, A tiles staged memory->LDS via GLOBAL_LOAD_ASYNC_TO_LDS_B128
// (ASYNCcnt), triple-buffered with one tile always in flight.
template <int CMODE, int OMODE, int KPAD>
__global__ __launch_bounds__(128)
void gemm_wmma_k(const unsigned short* __restrict__ Abf,
                 const unsigned short* __restrict__ Wt,
                 const float* __restrict__ Cin,
                 float* __restrict__ outF,
                 unsigned short* __restrict__ outB) {
  __shared__ __align__(16) unsigned short Atile[3][16][64];   // 3 x 2KB

  const int lane = threadIdx.x & 31;
  const int wave = threadIdx.x >> 5;
  const int ln   = lane & 15;
  const int hi   = lane >> 4;
  const int m0   = blockIdx.x * 16;
  const int n0a  = (blockIdx.y * 8 + wave * 2) * 16;   // tiles 0..23 (NPAD=384)
  const int gna  = n0a + ln, gnb = gna + 16;

  // accumulator init (C/D layout: VGPR r -> row hi*8+r, col ln)
  FAcc acc0, acc1;
#pragma unroll
  for (int r = 0; r < 8; ++r) {
    int gm = m0 + hi * 8 + r;
    float c0 = 0.0f, c1 = 0.0f;
    if (CMODE == 1) {
      if (gna < HID) c0 = Cin[(size_t)gm * HID + gna];
      if (gnb < HID) c1 = Cin[(size_t)gm * HID + gnb];
    } else if (CMODE == 2) {
      if (gna < HID) c0 = Cin[gna];
      if (gnb < HID) c1 = Cin[gnb];
    }
    acc0.f[r] = c0;
    acc1.f[r] = c1;
  }

  const int tm = threadIdx.x >> 3;          // 0..15 tile row
  const int tk = (threadIdx.x & 7) * 8;     // 8 consecutive bf16 per thread
  const unsigned short* arow = Abf + (size_t)(m0 + tm) * KPAD + tk;
  const unsigned short* bpa  = Wt + (size_t)gna * KPAD + hi * 8;
  const unsigned short* bpb  = Wt + (size_t)gnb * KPAD + hi * 8;

  // Low 32 bits of a flat LDS pointer == LDS byte offset (ISA 10.2 aperture map)
  const unsigned lds0 = (unsigned)(size_t)&Atile[0][tm][tk];

  constexpr int KITERS = KPAD / 64;

  // issue async DMA of K-tile `it` into LDS buffer it%3 (16B per lane);
  // with full unroll both operands are loop-invariant bases + immediates.
  auto issueA = [&](int it) {
    asm volatile("global_load_async_to_lds_b128 %0, %1, off"
                 :: "v"(lds0 + (unsigned)((it % 3) * 2048)),
                    "v"(arow + (size_t)it * 64)
                 : "memory");
  };

  issueA(0);
#pragma unroll
  for (int it = 0; it < KITERS; ++it) {
    if (it + 1 < KITERS) {
      issueA(it + 1);                                   // keep 1 tile in flight
      asm volatile("s_wait_asynccnt 0x1" ::: "memory"); // current tile landed
    } else {
      asm volatile("s_wait_asynccnt 0x0" ::: "memory");
    }
    __syncthreads();

    const unsigned short* At = &Atile[it % 3][0][0];
    BFrag a0, a1, b00, b01, b10, b11;
    // A frags (ISA 7.12.2): lane -> row ln; K runs {hi*8..+7},{16+hi*8..+7}
    a0.q[0] = *(const uint4*)(At + ln * 64 + hi * 8);
    a0.q[1] = *(const uint4*)(At + ln * 64 + 16 + hi * 8);
    a1.q[0] = *(const uint4*)(At + ln * 64 + 32 + hi * 8);
    a1.q[1] = *(const uint4*)(At + ln * 64 + 48 + hi * 8);
    // B frags: lane -> col ln; K runs {hi*8..+7},{16+hi*8..+7} (N-major packed weights)
    const unsigned short* p0 = bpa + it * 64;
    const unsigned short* p1 = bpb + it * 64;
    b00.q[0] = *(const uint4*)(p0);       b00.q[1] = *(const uint4*)(p0 + 16);
    b01.q[0] = *(const uint4*)(p0 + 32);  b01.q[1] = *(const uint4*)(p0 + 48);
    b10.q[0] = *(const uint4*)(p1);       b10.q[1] = *(const uint4*)(p1 + 16);
    b11.q[0] = *(const uint4*)(p1 + 32);  b11.q[1] = *(const uint4*)(p1 + 48);
    if (it + 1 < KITERS) {
      __builtin_prefetch(p0 + 64, 0, 3);
      __builtin_prefetch(p1 + 64, 0, 3);
    }

    acc0.v = __builtin_amdgcn_wmma_f32_16x16x32_bf16(
        false, a0.v, false, b00.v, (short)0, acc0.v, false, false);
    acc0.v = __builtin_amdgcn_wmma_f32_16x16x32_bf16(
        false, a1.v, false, b01.v, (short)0, acc0.v, false, false);
    acc1.v = __builtin_amdgcn_wmma_f32_16x16x32_bf16(
        false, a0.v, false, b10.v, (short)0, acc1.v, false, false);
    acc1.v = __builtin_amdgcn_wmma_f32_16x16x32_bf16(
        false, a1.v, false, b11.v, (short)0, acc1.v, false, false);
    // single barrier per iter: writer of buf (it+1)%3 races only with readers
    // from iteration it-2, which retired before the previous barrier.
  }

  // epilogue: ReLU + stores (only cols < 300 are real)
#pragma unroll
  for (int r = 0; r < 8; ++r) {
    int gm = m0 + hi * 8 + r;
    size_t ro = (size_t)gm * HID;
    float v0 = acc0.f[r], v1 = acc1.f[r];
    if (gna < HID) {
      float rv = fmaxf(v0, 0.0f);
      if (OMODE == 0)      { outF[ro + gna] = v0; outB[ro + gna] = f2bf(rv); }
      else if (OMODE == 1) { outB[ro + gna] = f2bf(rv); }
      else                 { outF[ro + gna] = rv; }
    }
    if (gnb < HID) {
      float rv = fmaxf(v1, 0.0f);
      if (OMODE == 0)      { outF[ro + gnb] = v1; outB[ro + gnb] = f2bf(rv); }
      else if (OMODE == 1) { outB[ro + gnb] = f2bf(rv); }
      else                 { outF[ro + gnb] = rv; }
    }
  }
}

// ---- readout ----
__global__ void count_k(const int* __restrict__ a2mol, float* __restrict__ counts) {
  int a = blockIdx.x * 256 + threadIdx.x;
  if (a < NB_ATOMS) atomicAdd(&counts[a2mol[a]], 1.0f);
}
__global__ void scatter_k(const float* __restrict__ hid,
                          const int* __restrict__ a2mol,
                          float* __restrict__ out) {
  long long idx = (long long)blockIdx.x * 256 + threadIdx.x;
  if (idx >= (long long)NB_ATOMS * HID) return;
  int a = (int)(idx / HID), h = (int)(idx % HID);
  atomicAdd(&out[(size_t)a2mol[a] * HID + h], hid[idx]);
}
__global__ void divide_k(float* __restrict__ out, const float* __restrict__ counts) {
  int i = blockIdx.x * 256 + threadIdx.x;
  if (i >= NB_MOLS * HID) return;
  out[i] /= fmaxf(counts[i / HID], 1.0f);
}

extern "C" void kernel_launch(void* const* d_in, const int* in_sizes, int n_in,
                              void* d_out, int out_size, void* d_ws, size_t ws_size,
                              hipStream_t stream) {
  const float* f_atoms = (const float*)d_in[0];
  const float* f_bonds = (const float*)d_in[1];
  const float* W_i     = (const float*)d_in[2];
  const float* W_h     = (const float*)d_in[3];
  const float* W_o     = (const float*)d_in[4];
  const float* b_o     = (const float*)d_in[5];
  const int*   a2b     = (const int*)d_in[6];
  const int*   b2a     = (const int*)d_in[7];
  const int*   b2revb  = (const int*)d_in[8];
  const int*   a2mol   = (const int*)d_in[9];
  float* out = (float*)d_out;

  char* ws = (char*)d_ws;
  size_t off = 0;
  auto carve = [&](size_t bytes) {
    char* p = ws + off;
    off = (off + bytes + 255) & ~(size_t)255;
    return p;
  };
  float*          inp   = (float*)carve((size_t)NB_BONDS * HID * 4);
  unsigned short* msg0  = (unsigned short*)carve((size_t)NB_BONDS * HID * 2);
  unsigned short* msg1  = (unsigned short*)carve((size_t)NB_BONDS * HID * 2);
  float*          amsg  = (float*)carve((size_t)NB_ATOMS * HID * 4);
  float*          hid   = (float*)carve((size_t)NB_ATOMS * HID * 4);
  float*          cnts  = (float*)carve((size_t)NB_MOLS * 4);
  unsigned short* Abond = (unsigned short*)carve((size_t)NB_BONDS * KP_I * 2);
  unsigned short* Anew  = (unsigned short*)carve((size_t)NB_BONDS * KP_H * 2);
  unsigned short* Aatom = Anew;  // alias: atom A-matrix built after last Anew use
  unsigned short* WiT   = (unsigned short*)carve((size_t)NPAD * KP_I * 2);
  unsigned short* WhT   = (unsigned short*)carve((size_t)NPAD * KP_H * 2);
  unsigned short* WoT   = (unsigned short*)carve((size_t)NPAD * KP_O * 2);

  fill_zero_k<<<(NB_MOLS * HID + 255) / 256, 256, 0, stream>>>(out, NB_MOLS * HID);
  fill_zero_k<<<(NB_MOLS + 255) / 256, 256, 0, stream>>>(cnts, NB_MOLS);

  pack_w_k<<<(NPAD * KP_I + 255) / 256, 256, 0, stream>>>(W_i, BOND_FDIM, HID, WiT, KP_I);
  pack_w_k<<<(NPAD * KP_H + 255) / 256, 256, 0, stream>>>(W_h, HID, HID, WhT, KP_H);
  pack_w_k<<<(NPAD * KP_O + 255) / 256, 256, 0, stream>>>(W_o, ATOM_FDIM + HID, HID, WoT, KP_O);

  dim3 gb(NB_BONDS / 16, 3);   // 25000 x 3 (8 N-tiles per block, 24 >= 19)
  dim3 ga(NB_ATOMS / 16, 3);

  // GEMM 1: inp = f_bonds @ W_i ; msg0 = relu(inp)
  prep_bonds_k<<<NB_BONDS, 96, 0, stream>>>(f_bonds, Abond);
  gemm_wmma_k<0, 0, KP_I><<<gb, 128, 0, stream>>>(Abond, WiT, nullptr, inp, msg0);

  // depth 1: msg1 = relu(inp + (amsg[b2a]-msg0[b2revb]) @ W_h)
  amsg_k<<<NB_ATOMS, 160, 0, stream>>>(msg0, a2b, amsg);
  prep_newmsg_k<<<NB_BONDS, 160, 0, stream>>>(amsg, msg0, b2a, b2revb, Anew);
  gemm_wmma_k<1, 1, KP_H><<<gb, 128, 0, stream>>>(Anew, WhT, inp, nullptr, msg1);

  // depth 2: msg0 = relu(inp + (amsg[b2a]-msg1[b2revb]) @ W_h)
  amsg_k<<<NB_ATOMS, 160, 0, stream>>>(msg1, a2b, amsg);
  prep_newmsg_k<<<NB_BONDS, 160, 0, stream>>>(amsg, msg1, b2a, b2revb, Anew);
  gemm_wmma_k<1, 1, KP_H><<<gb, 128, 0, stream>>>(Anew, WhT, inp, nullptr, msg0);

  // output stage: hid = relu([f_atoms | amsg] @ W_o + b_o)
  amsg_k<<<NB_ATOMS, 160, 0, stream>>>(msg0, a2b, amsg);
  prep_atoms_k<<<NB_ATOMS, 224, 0, stream>>>(f_atoms, amsg, Aatom);
  gemm_wmma_k<2, 2, KP_O><<<ga, 128, 0, stream>>>(Aatom, WoT, b_o, hid, nullptr);

  // segment mean readout
  count_k<<<(NB_ATOMS + 255) / 256, 256, 0, stream>>>(a2mol, cnts);
  long long nel = (long long)NB_ATOMS * HID;
  scatter_k<<<(unsigned)((nel + 255) / 256), 256, 0, stream>>>(hid, a2mol, out);
  divide_k<<<(NB_MOLS * HID + 255) / 256, 256, 0, stream>>>(out, cnts);
}
